// GLCM_867583394638
// MI455X (gfx1250) — matrix operand order repaired and checked by reading, not compiled, not run on hardware
//
#include <hip/hip_runtime.h>

typedef __attribute__((ext_vector_type(16))) _Float16 v16h;
typedef __attribute__((ext_vector_type(8)))  float    v8f;
typedef __attribute__((ext_vector_type(4)))  unsigned int u32x4;
typedef __attribute__((ext_vector_type(8)))  int      i32x8;
typedef __attribute__((ext_vector_type(4)))  int      i32x4;

// Problem sizes (fixed by the reference)
constexpr int B_SZ   = 8;
constexpr int M_PIX  = 51529;                      // 227*227
constexpr int NQ     = 256;                        // quant levels
constexpr int OUTF   = 32;
constexpr int GL     = NQ * NQ;                    // 65536 glcm elems / batch
constexpr int CHUNK  = 32;                         // K per WMMA
constexpr int TOT_CHUNKS   = (M_PIX + CHUNK - 1) / CHUNK;   // 1611
constexpr int STAGE_CHUNKS = 16;
constexpr int STAGE_F      = STAGE_CHUNKS * CHUNK;          // 512 floats staged
constexpr int STAGE_PAD    = STAGE_F + 8;                   // 520 floats (16B aligned stride)
constexpr int TILE_ELEMS   = STAGE_F + 1;                   // 513 (extra for superdiag diff)
constexpr int MAX_SEG      = 16;

#if defined(__has_builtin)
# if __has_builtin(__builtin_amdgcn_tensor_load_to_lds) && \
     __has_builtin(__builtin_amdgcn_s_wait_tensorcnt)
#  define USE_TDM 1
# endif
#endif

union Pack16 { unsigned int u[8]; v16h v; };

__device__ __forceinline__ unsigned int pk_sat(float x0, float x1) {
    // clamp to [0,1] (compiler fuses into v_sub clamp), pack 2xf32 -> packed f16
    float c0 = __builtin_amdgcn_fmed3f(x0, 0.0f, 1.0f);
    float c1 = __builtin_amdgcn_fmed3f(x1, 0.0f, 1.0f);
#if __has_builtin(__builtin_amdgcn_cvt_pkrtz)
    return __builtin_bit_cast(unsigned int, __builtin_amdgcn_cvt_pkrtz(c0, c1));
#else
    union { _Float16 h[2]; unsigned int u; } p;
    p.h[0] = (_Float16)c0; p.h[1] = (_Float16)c1;
    return p.u;
#endif
}

#ifdef USE_TDM
// Issue a TDM DMA: copy TILE_ELEMS f32 from gbase+kbase into LDS byte offset
// lds_off. tensor_dim0 = remaining elements, so OOB reads return 0 -> exactly
// the zero padding the GLCM tail needs. D# layout per CDNA5 ISA 8.3/8.4.
__device__ __forceinline__ void tdm_issue(const float* gbase, int kbase,
                                          unsigned lds_off) {
    unsigned long long ga = (unsigned long long)(size_t)(gbase + kbase);
    unsigned td0 = (unsigned)(M_PIX - kbase);       // remaining valid elements
    u32x4 g0;
    g0[0] = 1u;                                     // count=1, is_restore=0
    g0[1] = lds_off;                                // lds_addr (bytes)
    g0[2] = (unsigned)ga;                           // global_addr[31:0]
    g0[3] = (unsigned)((ga >> 32) & 0x01FFFFFFu)    // global_addr[56:32]
            | 0x80000000u;                          // type=2 ("image")
    i32x8 g1;
    g1[0] = 0x00020000;                             // data_size=4B, wg_mask=0
    g1[1] = (int)((td0 & 0xFFFFu) << 16);           // tensor_dim0[15:0]
    g1[2] = (int)(((td0 >> 16) & 0xFFFFu)           // tensor_dim0[31:16]
            | (1u << 16));                          // tensor_dim1 = 1
    g1[3] = (int)((unsigned)TILE_ELEMS << 16);      // tile_dim0 = 513
    g1[4] = 0;                                      // tile_dim1/2 unused
    g1[5] = (int)td0;                               // tensor_dim0_stride lo
    g1[6] = 0;
    g1[7] = 0;
    i32x4 z4 = (i32x4)0;
#if __clang_major__ >= 23
    i32x8 z8 = (i32x8)0;
    __builtin_amdgcn_tensor_load_to_lds(g0, g1, z4, z4, z8, 0);
#else
    __builtin_amdgcn_tensor_load_to_lds(g0, g1, z4, z4, 0);
#endif
}
#endif

// Kernel 1: partial GLCM via f16 WMMA with f32 accumulate.
// grid = (8 tiles, nseg, 8 batches), block = 128 (4 waves).
// Each WG computes a 128x64 tile of one batch's 256x256 GLCM over one K segment.
// Each wave: 64x32 patch = 4x2 WMMA subtiles. Raw pixel chunks are staged in
// LDS (TDM double-buffered when available) and fragments built on the fly.
__global__ __launch_bounds__(128)
void glcm_wmma_kernel(const float* __restrict__ x,
                      const float* __restrict__ phiA,
                      const float* __restrict__ phiB,
                      float* __restrict__ part,
                      int chunks_per_seg)
{
    __shared__ float lds[2][STAGE_PAD];

    const int b    = blockIdx.z;
    const int seg  = blockIdx.y;
    const int tile = blockIdx.x;          // 0..7
    const int ti   = tile & 1;            // row block of 128
    const int tj   = tile >> 1;           // col block of 64 (0..3)

    const int tid  = threadIdx.x;
    const int wave = tid >> 5;
    const int lane = tid & 31;
    const int n    = lane & 15;
    const int half = lane >> 4;

    const int wi = wave & 1;              // +0 / +64 rows
    const int wj = wave >> 1;             // +0 / +32 cols
    const int i_wave = ti * 128 + wi * 64;
    const int j_wave = tj * 64  + wj * 32;

    // LDS buffer addresses escape into asm so the compiler must assume the
    // buffers are written (TDM writes are invisible to alias analysis).
    asm volatile("" : : "r"(&lds[0][0]), "r"(&lds[1][0]) : "memory");

    // per-lane thresholds (loaded once)
    float pA[4], pB[2];
#pragma unroll
    for (int s = 0; s < 4; ++s) pA[s] = phiA[i_wave + 16 * s + n];
#pragma unroll
    for (int u = 0; u < 2; ++u) pB[u] = phiB[j_wave + 16 * u + n];

    v8f acc[4][2];
#pragma unroll
    for (int s = 0; s < 4; ++s)
#pragma unroll
        for (int u = 0; u < 2; ++u) acc[s][u] = (v8f)(0.0f);

    const float* a = x + (size_t)b * M_PIX;

    const int c0 = seg * chunks_per_seg;
    int c1 = c0 + chunks_per_seg; if (c1 > TOT_CHUNKS) c1 = TOT_CHUNKS;
    const int nstages = (c1 > c0) ? (c1 - c0 + STAGE_CHUNKS - 1) / STAGE_CHUNKS : 0;

#ifdef USE_TDM
    const unsigned lds_off0 = (unsigned)(size_t)(void*)&lds[0][0];
    const unsigned lds_off1 = (unsigned)(size_t)(void*)&lds[1][0];
    if (wave == 0 && nstages > 0) tdm_issue(a, c0 * CHUNK, lds_off0);
#endif

    for (int st = 0; st < nstages; ++st) {
        const int blk    = c0 + st * STAGE_CHUNKS;
        const int cur    = st & 1;
        const int nchunk = (c1 - blk < STAGE_CHUNKS) ? (c1 - blk) : STAGE_CHUNKS;
        const float* L   = lds[cur];

#ifdef USE_TDM
        if (wave == 0) {
            if (st + 1 < nstages) {
                // issue next stage into the other buffer, keep newest in flight
                tdm_issue(a, (blk + STAGE_CHUNKS) * CHUNK, cur ? lds_off0 : lds_off1);
                __builtin_amdgcn_s_wait_tensorcnt(1);
            } else {
                __builtin_amdgcn_s_wait_tensorcnt(0);
            }
        }
        __syncthreads();                  // release compute on buffer `cur`
#else
        const int kbase = blk * CHUNK;
        const int nfl   = nchunk * CHUNK + 1;   // +1 float for superdiag diff
        __syncthreads();                  // protect previous readers
        for (int t = tid; t < nfl; t += 128) {
            const int g = kbase + t;
            lds[cur][t] = (g < M_PIX) ? a[g] : 0.0f;
        }
        __syncthreads();
        {   // prefetch next stage block (global_prefetch_b8)
            const int nk = kbase + STAGE_F + tid * 4;
            if (nk < M_PIX && (blk + STAGE_CHUNKS) < c1) __builtin_prefetch(&a[nk]);
        }
#endif

#pragma unroll 1
        for (int cc = 0; cc < nchunk; ++cc) {
            const int base = cc * CHUNK;

            // raw values for B fragments: 17 contiguous floats at 16*half
            float cB[17];
#pragma unroll
            for (int e = 0; e < 17; ++e) cB[e] = L[base + 16 * half + e];
            // raw values for A fragments: two 8-float runs (shared by all 4 subtiles)
            float cA[16];
#pragma unroll
            for (int e = 0; e < 8; ++e) {
                cA[e]     = L[base + 16 * half + e];
                cA[8 + e] = L[base + 16 * half + 8 + e];
            }
            // superdiagonal differences b_k = a_k - a_{k+1}
            float d[16];
#pragma unroll
            for (int e = 0; e < 16; ++e) d[e] = cB[e] - cB[e + 1];

            v16h Af[4], Bf[2];
#pragma unroll
            for (int s = 0; s < 4; ++s) {
                Pack16 m;
#pragma unroll
                for (int e = 0; e < 16; e += 2)
                    m.u[e >> 1] = pk_sat(cA[e] - pA[s], cA[e + 1] - pA[s]);
                Af[s] = m.v;
            }
#pragma unroll
            for (int u = 0; u < 2; ++u) {
                Pack16 m;
#pragma unroll
                for (int e = 0; e < 16; e += 2)
                    m.u[e >> 1] = pk_sat(d[e] - pB[u], d[e + 1] - pB[u]);
                Bf[u] = m.v;
            }

#pragma unroll
            for (int s = 0; s < 4; ++s)
#pragma unroll
                for (int u = 0; u < 2; ++u)
                    acc[s][u] = __builtin_amdgcn_wmma_f32_16x16x32_f16(
                        false, Af[s], false, Bf[u], (short)0, acc[s][u],
                        false, false);
        }

#ifdef USE_TDM
        __syncthreads();                  // all readers done before buffer reuse
#endif
    }

    // write f32 partials; D layout: VGPR r, lane l -> row = 8*(l>>4)+r, col = l&15
    float* dst = part + (size_t)(seg * B_SZ + b) * GL;
#pragma unroll
    for (int s = 0; s < 4; ++s)
#pragma unroll
        for (int u = 0; u < 2; ++u) {
            const int col = j_wave + 16 * u + n;
#pragma unroll
            for (int r = 0; r < 8; ++r) {
                const int row = i_wave + 16 * s + 8 * half + r;
                dst[(size_t)row * NQ + col] = acc[s][u][r];
            }
        }
}

// Kernel 2: out[b][o] = relu(bias[o] + sum_k (sum_s part[s][b][k]) * w[k][o])
// grid = 8 (batch), block = 1024 (32 waves); deterministic shuffle-tree reduce.
__global__ __launch_bounds__(1024)
void glcm_head_kernel(const float* __restrict__ part, int nseg,
                      const float* __restrict__ w,
                      const float* __restrict__ bias,
                      float* __restrict__ out)
{
    const int b   = blockIdx.x;
    const int tid = threadIdx.x;

    float acc[OUTF];
#pragma unroll
    for (int o = 0; o < OUTF; ++o) acc[o] = 0.0f;

    for (int k = tid; k < GL; k += 1024) {
        float g = 0.0f;
        for (int s = 0; s < nseg; ++s)
            g += part[(size_t)(s * B_SZ + b) * GL + k];
        const float4* wr = (const float4*)(w + (size_t)k * OUTF);
#pragma unroll
        for (int q = 0; q < 8; ++q) {
            const float4 wv = wr[q];
            acc[4 * q + 0] += g * wv.x;
            acc[4 * q + 1] += g * wv.y;
            acc[4 * q + 2] += g * wv.z;
            acc[4 * q + 3] += g * wv.w;
        }
    }

    // wave32 tree reduction
#pragma unroll
    for (int o = 0; o < OUTF; ++o)
        for (int off = 16; off > 0; off >>= 1)
            acc[o] += __shfl_down(acc[o], off, 32);

    __shared__ float red[32][OUTF];
    const int wid = tid >> 5, lane = tid & 31;
    if (lane == 0) {
#pragma unroll
        for (int o = 0; o < OUTF; ++o) red[wid][o] = acc[o];
    }
    __syncthreads();
    if (tid < OUTF) {
        float s = bias[tid];
        for (int wv = 0; wv < 32; ++wv) s += red[wv][tid];
        out[b * OUTF + tid] = fmaxf(s, 0.0f);
    }
}

extern "C" void kernel_launch(void* const* d_in, const int* in_sizes, int n_in,
                              void* d_out, int out_size, void* d_ws, size_t ws_size,
                              hipStream_t stream) {
    (void)in_sizes; (void)n_in; (void)out_size;
    const float* x    = (const float*)d_in[0];
    const float* phiA = (const float*)d_in[1];
    const float* phiB = (const float*)d_in[2];
    const float* w    = (const float*)d_in[3];
    const float* bias = (const float*)d_in[4];
    float* out  = (float*)d_out;
    float* part = (float*)d_ws;

    // K-split factor chosen deterministically from workspace size (2 MB per segment)
    const size_t seg_bytes = (size_t)B_SZ * GL * sizeof(float);
    int nseg = (int)(ws_size / seg_bytes);
    if (nseg < 1) nseg = 1;
    if (nseg > MAX_SEG) nseg = MAX_SEG;
    const int cps = (TOT_CHUNKS + nseg - 1) / nseg;

    dim3 g1(8, nseg, B_SZ);
    glcm_wmma_kernel<<<g1, 128, 0, stream>>>(x, phiA, phiB, part, cps);
    glcm_head_kernel<<<dim3(B_SZ), 1024, 0, stream>>>(part, nseg, w, bias, out);
}
